// ContextualAttentionModule_4922032521651
// MI455X (gfx1250) — matrix operand order repaired and checked by reading, not compiled, not run on hardware
//
#include <hip/hip_runtime.h>
#include <hip/hip_bf16.h>

// ---------------------------------------------------------------------------
// ContextualAttention on MI455X (gfx1250): two 4096x{4096,1152}x{1152,4096}
// bf16 GEMMs on v_wmma_f32_16x16x32_bf16 with async global->LDS double
// buffering (ASYNCcnt DMA), plus fused box-sum/softmax and fold.
// ---------------------------------------------------------------------------

typedef __attribute__((ext_vector_type(16))) __bf16        v16bf;
typedef __attribute__((ext_vector_type(8)))  float         v8f;
typedef __attribute__((ext_vector_type(4)))  unsigned int  u32x4;

union FragBF {            // 32 bytes: one WMMA A/B operand per lane
    u32x4 u[2];
    v16bf v;
};

__device__ inline unsigned short f2bf(float f) {
    union { float f; unsigned int u; } v; v.f = f;
    unsigned int u = v.u;
    u += 0x7FFFu + ((u >> 16) & 1u);   // round-to-nearest-even
    return (unsigned short)(u >> 16);
}

// per-lane 16B DMA: global -> LDS, tracked by ASYNCcnt
__device__ inline void async_cp16(const unsigned short* gsrc,
                                  const unsigned short* ldst) {
    unsigned laddr = (unsigned)(unsigned long long)(const void*)ldst;
    unsigned long long gaddr = (unsigned long long)(const void*)gsrc;
    asm volatile("global_load_async_to_lds_b128 %0, %1, off"
                 :: "v"(laddr), "v"(gaddr) : "memory");
}

#define CCH 128
#define HH  64
#define WW  64
#define LL  4096          // H*W
#define KD  1152          // C*9

// ---------------------------------------------------------------------------
// Kernel 1: im2col + EPS + patch L2-norm.  One block per patch location l.
//   Afg[l][s*C+c] = fg patch (zero pad)          (bf16)
//   Kn [l][s*C+c] = (bg patch + eps)/||.||       (bf16)
//   KnT[s*C+c][l] = transpose of Kn              (bf16, for GEMM2)
// ---------------------------------------------------------------------------
__global__ __launch_bounds__(128) void im2col_norm_kernel(
    const float* __restrict__ fg, const float* __restrict__ mask,
    unsigned short* __restrict__ Afg, unsigned short* __restrict__ Kn,
    unsigned short* __restrict__ KnT)
{
    int l = blockIdx.x;
    int c = threadIdx.x;
    int h = l >> 6, w = l & 63;

    float vbg[9], vfg[9];
    float ss = 0.0f;
#pragma unroll
    for (int s = 0; s < 9; ++s) {
        int e = s / 3, f = s % 3;
        int y = h + e - 1, x = w + f - 1;
        bool inb = (y >= 0) && (y < HH) && (x >= 0) && (x < WW);
        float vf = 0.0f, mv = 0.0f;
        if (inb) {
            vf = fg[c * (HH * WW) + y * WW + x];
            mv = mask[y * WW + x];
        }
        float vb = (inb ? vf * (1.0f - mv) : 0.0f) + 1e-7f;
        vfg[s] = inb ? vf : 0.0f;
        vbg[s] = vb;
        ss += vb * vb;
    }

    __shared__ float red[128];
    red[c] = ss;
    __syncthreads();
    for (int off = 64; off > 0; off >>= 1) {
        if (c < off) red[c] += red[c + off];
        __syncthreads();
    }
    float inv = rsqrtf(red[0]);

#pragma unroll
    for (int s = 0; s < 9; ++s) {
        int kidx = s * CCH + c;
        unsigned short kb = f2bf(vbg[s] * inv);
        Kn [(size_t)l * KD + kidx] = kb;
        KnT[(size_t)kidx * LL + l] = kb;
        Afg[(size_t)l * KD + kidx] = f2bf(vfg[s]);
    }
}

// ---------------------------------------------------------------------------
// Kernel 2: C[M][N](f32) = A[M][Kd](bf16,row-major) * B[N][Kd](bf16)^T
// Block tile (MT*64) x 128, 256 threads = 8 waves (wave32) in a 4x2 grid;
// wave tile (MT*16) x 64 = MT x 4 WMMA f32 16x16 accumulators.
// K stepped by 32 with double-buffered async global->LDS DMA (ASYNCcnt).
//   MT=4: 256x128 block, 512 B LDS per WMMA (GEMM1, M=N=4096)
//   MT=2: 128x128 block (GEMM2, N=1152 -> more blocks for occupancy)
// ---------------------------------------------------------------------------
#define BN 128
#define BK 32
#define LSTR 40           // LDS row stride in bf16 elems (32 + 8 pad, 80B rows)

template <int MT>
__global__ __launch_bounds__(256) void gemm_bf16_nt_kernel(
    const unsigned short* __restrict__ A,
    const unsigned short* __restrict__ B,
    float* __restrict__ C, int M, int N, int Kd)
{
    __shared__ __align__(16) unsigned short As[2][MT * 64 * LSTR];
    __shared__ __align__(16) unsigned short Bs[2][BN * LSTR];

    const int tid  = threadIdx.x;
    const int m0   = blockIdx.y * (MT * 64);
    const int n0   = blockIdx.x * BN;
    const int wave = tid >> 5;
    const int lane = tid & 31;
    const int lm   = lane & 15;            // row/col within 16-wide tile
    const int hs   = lane >> 4;            // K-half select (ISA VGPR layout)
    const int wm   = (wave & 3) * (MT * 16); // wave M offset in block
    const int wn   = (wave >> 2) * 64;       // wave N offset in block

    // cooperative loader: 4 threads cover one 64B row-chunk
    const int lk = (tid & 3) * 8;     // element offset 0/8/16/24 (16B chunks)
    const int lr = tid >> 2;          // rows 0..63

    v8f acc[MT][4] = {};              // zero-init accumulators

    const int T = Kd / BK;

    // issue tile: MT+2 async 16B copies per thread (MT A rows + 2 B rows)
    auto issue_tile = [&](int t, int buf) {
        int k0 = t * BK;
#pragma unroll
        for (int rr = 0; rr < MT; ++rr) {
            int row = lr + rr * 64;
            async_cp16(A + (size_t)(m0 + row) * Kd + k0 + lk,
                       &As[buf][row * LSTR + lk]);
        }
#pragma unroll
        for (int rr = 0; rr < 2; ++rr) {
            int row = lr + rr * 64;
            async_cp16(B + (size_t)(n0 + row) * Kd + k0 + lk,
                       &Bs[buf][row * LSTR + lk]);
        }
    };

    issue_tile(0, 0);                  // prologue

    for (int t = 0; t < T; ++t) {
        int cur = t & 1;
        if (t + 1 < T) {
            issue_tile(t + 1, cur ^ 1);            // prefetch next tile
            if constexpr (MT == 4)
                asm volatile("s_wait_asynccnt 0x6" ::: "memory");
            else
                asm volatile("s_wait_asynccnt 0x4" ::: "memory");
        } else {
            asm volatile("s_wait_asynccnt 0x0" ::: "memory");
        }
        __syncthreads();               // all waves' current tile is in LDS

        FragBF fa[MT], fb[4];
#pragma unroll
        for (int i = 0; i < MT; ++i) {
            // A 16x32 bf16 frag: lane<16 -> K {0..7,16..23}, lane>=16 -> +8
            const unsigned short* base = &As[cur][(wm + i * 16 + lm) * LSTR];
            fa[i].u[0] = *(const u32x4*)(base + hs * 8);
            fa[i].u[1] = *(const u32x4*)(base + 16 + hs * 8);
        }
#pragma unroll
        for (int j = 0; j < 4; ++j) {
            // B 32x16 bf16 frag: lane = N col; lane<16 -> K {0..7,16..23}, lane>=16 -> +8
            const unsigned short* base =
                &Bs[cur][(wn + j * 16 + lm) * LSTR];
            fb[j].u[0] = *(const u32x4*)(base + hs * 8);
            fb[j].u[1] = *(const u32x4*)(base + 16 + hs * 8);
        }
#pragma unroll
        for (int i = 0; i < MT; ++i)
#pragma unroll
            for (int j = 0; j < 4; ++j)
                acc[i][j] = __builtin_amdgcn_wmma_f32_16x16x32_bf16(
                    false, fa[i].v, false, fb[j].v,
                    (short)0, acc[i][j], false, false);
        __syncthreads();               // done reading cur; safe to overwrite
    }

    // D layout: lanes 0-15 -> N=lane, M=vgpr r; lanes 16-31 -> M = 8+r
#pragma unroll
    for (int i = 0; i < MT; ++i)
#pragma unroll
        for (int j = 0; j < 4; ++j) {
            int n = n0 + wn + j * 16 + lm;
#pragma unroll
            for (int r = 0; r < 8; ++r) {
                int m = m0 + wm + i * 16 + hs * 8 + r;
                C[(size_t)m * N + n] = acc[i][j][r];
            }
        }
}

// ---------------------------------------------------------------------------
// Kernel 3: fused 3x3 spatial box-sum over p + softmax over l, bf16 output.
// One block (256 thr) per p; each thread owns 16 strided columns of row p.
// ---------------------------------------------------------------------------
__global__ __launch_bounds__(256) void box_softmax_kernel(
    const float* __restrict__ S, unsigned short* __restrict__ attn)
{
    int p = blockIdx.x;
    int h = p >> 6, w = p & 63;
    int tid = threadIdx.x;

    float acc[16];
#pragma unroll
    for (int i = 0; i < 16; ++i) acc[i] = 0.0f;

    for (int dh = -1; dh <= 1; ++dh) {
        int y = h + dh; if (y < 0 || y >= HH) continue;
        for (int dw = -1; dw <= 1; ++dw) {
            int x = w + dw; if (x < 0 || x >= WW) continue;
            const float* row = S + (size_t)(y * WW + x) * LL;
#pragma unroll
            for (int i = 0; i < 16; ++i) acc[i] += row[tid + i * 256];
        }
    }

    __shared__ float sred[256];
    float mx = -3.4e38f;
#pragma unroll
    for (int i = 0; i < 16; ++i) mx = fmaxf(mx, acc[i]);
    sred[tid] = mx;
    __syncthreads();
    for (int off = 128; off > 0; off >>= 1) {
        if (tid < off) sred[tid] = fmaxf(sred[tid], sred[tid + off]);
        __syncthreads();
    }
    float MX = sred[0];
    __syncthreads();

    float sum = 0.0f;
#pragma unroll
    for (int i = 0; i < 16; ++i) {
        acc[i] = __expf(acc[i] - MX);
        sum += acc[i];
    }
    sred[tid] = sum;
    __syncthreads();
    for (int off = 128; off > 0; off >>= 1) {
        if (tid < off) sred[tid] += sred[tid + off];
        __syncthreads();
    }
    float inv = 1.0f / sred[0];

    unsigned short* orow = attn + (size_t)p * LL;
#pragma unroll
    for (int i = 0; i < 16; ++i)
        orow[tid + i * 256] = f2bf(acc[i] * inv);
}

// ---------------------------------------------------------------------------
// Kernel 4: col2im fold of RC[L][KD] + mask blend:
//   out[c,x] = (1/9)*m(x)*sum_{e,f} RC[x-(e-1,f-1)][(e*3+f)*C+c] + fg*(1-m)
// ---------------------------------------------------------------------------
__global__ __launch_bounds__(256) void fold_combine_kernel(
    const float* __restrict__ RC, const float* __restrict__ fg,
    const float* __restrict__ mask, float* __restrict__ out)
{
    int idx = blockIdx.x * blockDim.x + threadIdx.x;
    if (idx >= CCH * HH * WW) return;
    int c = idx >> 12;
    int pix = idx & 4095;
    int h = pix >> 6, w = pix & 63;

    float s = 0.0f;
#pragma unroll
    for (int e = 0; e < 3; ++e) {
        int y = h - (e - 1); if (y < 0 || y >= HH) continue;
#pragma unroll
        for (int f = 0; f < 3; ++f) {
            int x = w - (f - 1); if (x < 0 || x >= WW) continue;
            s += RC[(size_t)(y * WW + x) * KD + (e * 3 + f) * CCH + c];
        }
    }
    float m = mask[pix];
    out[idx] = s * m * (1.0f / 9.0f) + fg[idx] * (1.0f - m);
}

// ---------------------------------------------------------------------------
extern "C" void kernel_launch(void* const* d_in, const int* in_sizes, int n_in,
                              void* d_out, int out_size, void* d_ws, size_t ws_size,
                              hipStream_t stream)
{
    const int Bn = 4;
    const float* fg_all = (const float*)d_in[0];   // [4,128,64,64]
    const float* mk_all = (const float*)d_in[1];   // [4,1,64,64]
    float* out = (float*)d_out;

    char* ws = (char*)d_ws;
    const size_t szIm = (size_t)LL * KD * sizeof(unsigned short); // 9.44 MB
    unsigned short* Afg  = (unsigned short*)(ws);
    unsigned short* Kn   = (unsigned short*)(ws + szIm);
    unsigned short* KnT  = (unsigned short*)(ws + 2 * szIm);
    float*          S    = (float*)         (ws + 3 * szIm);
    unsigned short* attn = (unsigned short*)(ws + 3 * szIm + (size_t)LL * LL * 4);
    float*          RC   = (float*)         (ws + 3 * szIm + (size_t)LL * LL * 4
                                                           + (size_t)LL * LL * 2);

    for (int b = 0; b < Bn; ++b) {
        const float* fg = fg_all + (size_t)b * CCH * HH * WW;
        const float* mk = mk_all + (size_t)b * HH * WW;
        float* ob = out + (size_t)b * CCH * HH * WW;

        im2col_norm_kernel<<<LL, 128, 0, stream>>>(fg, mk, Afg, Kn, KnT);

        // S[p][l] = Afg . Kn^T : M=N=4096, K=1152; 256x128 blocks
        gemm_bf16_nt_kernel<4><<<dim3(LL / BN, LL / 256), 256, 0, stream>>>(
            Afg, Kn, S, LL, LL, KD);

        box_softmax_kernel<<<LL, 256, 0, stream>>>(S, attn);

        // RC[p][kidx] = attn . KnT^T : M=4096, N=1152, K=4096; 128x128 blocks
        gemm_bf16_nt_kernel<2><<<dim3(KD / BN, LL / 128), 256, 0, stream>>>(
            attn, KnT, RC, LL, KD, LL);

        fold_combine_kernel<<<(CCH * HH * WW + 255) / 256, 256, 0, stream>>>(
            RC, fg, mk, ob);
    }
}
